// GNNBlock_4071628996857
// MI455X (gfx1250) — compile-verified, matrix operand order checked by reading, not run on hardware
//
#include <hip/hip_runtime.h>
#include <hip/hip_bf16.h>

#define N_NODES 50000
#define E_EDGES 800000
#define DFEAT   128
#define LN_EPS  1e-5f

typedef __attribute__((ext_vector_type(8)))  __bf16 v8bf;
typedef __attribute__((ext_vector_type(16))) __bf16 v16bf;
typedef __attribute__((ext_vector_type(8)))  float  v8f;

static __device__ __forceinline__ unsigned short f32_to_bf16_rne(float f) {
    unsigned int u = __float_as_uint(f);
    unsigned int r = u + 0x7FFFu + ((u >> 16) & 1u);   // round-to-nearest-even
    return (unsigned short)(r >> 16);
}

// ---------------- Kernel 1: zero agg (N*D) + deg (N), laid out contiguously ----------------
__global__ void zero_f32(float* __restrict__ p, int n) {
    int i = blockIdx.x * blockDim.x + threadIdx.x;
    if (i < n) p[i] = 0.0f;
}

// ---------------- Kernel 2: convert f32 weights -> bf16 ----------------
__global__ void cvt_f32_bf16(const float* __restrict__ src, unsigned short* __restrict__ dst, int n) {
    int i = blockIdx.x * blockDim.x + threadIdx.x;
    if (i < n) dst[i] = f32_to_bf16_rne(src[i]);
}

// ---------------- Kernel 3: LayerNorm + ReLU + dropout-mask, one wave per row ----------------
__global__ void __launch_bounds__(256)
ln_relu_drop(const float* __restrict__ x, const float* __restrict__ mask,
             const float* __restrict__ gamma, const float* __restrict__ beta,
             float* __restrict__ h, unsigned short* __restrict__ hb) {
    int wave = (blockIdx.x * blockDim.x + threadIdx.x) >> 5;  // row
    int lane = threadIdx.x & 31;
    if (wave >= N_NODES) return;
    size_t base = (size_t)wave * DFEAT + (size_t)lane * 4;
    float4 xv = *(const float4*)(x + base);
    float s  = xv.x + xv.y + xv.z + xv.w;
    float ss = xv.x*xv.x + xv.y*xv.y + xv.z*xv.z + xv.w*xv.w;
    #pragma unroll
    for (int m = 16; m >= 1; m >>= 1) {
        s  += __shfl_xor(s,  m, 32);
        ss += __shfl_xor(ss, m, 32);
    }
    float mu   = s * (1.0f / DFEAT);
    float var  = ss * (1.0f / DFEAT) - mu * mu;
    float rstd = rsqrtf(var + LN_EPS);
    float4 g  = *(const float4*)(gamma + lane * 4);
    float4 b  = *(const float4*)(beta  + lane * 4);
    float4 mk = *(const float4*)(mask + base);
    float4 hv;
    hv.x = fmaxf((xv.x - mu) * rstd * g.x + b.x, 0.0f) * mk.x;
    hv.y = fmaxf((xv.y - mu) * rstd * g.y + b.y, 0.0f) * mk.y;
    hv.z = fmaxf((xv.z - mu) * rstd * g.z + b.z, 0.0f) * mk.z;
    hv.w = fmaxf((xv.w - mu) * rstd * g.w + b.w, 0.0f) * mk.w;
    *(float4*)(h + base) = hv;
    ushort4 hb4;
    hb4.x = f32_to_bf16_rne(hv.x); hb4.y = f32_to_bf16_rne(hv.y);
    hb4.z = f32_to_bf16_rne(hv.z); hb4.w = f32_to_bf16_rne(hv.w);
    *(ushort4*)(hb + base) = hb4;
}

// ---------------- Kernel 4: edge scatter, one wave per edge (coalesced float4 per lane) ----------------
__global__ void __launch_bounds__(256)
scatter_edges(const int* __restrict__ ei, const float* __restrict__ h,
              float* __restrict__ agg, float* __restrict__ deg) {
    int wave = (blockIdx.x * blockDim.x + threadIdx.x) >> 5;  // edge
    int lane = threadIdx.x & 31;
    if (wave >= E_EDGES) return;
    int src = ei[wave];
    int dst = ei[E_EDGES + wave];
    float4 m = *(const float4*)(h + (size_t)src * DFEAT + (size_t)lane * 4);
    float* a = agg + (size_t)dst * DFEAT + (size_t)lane * 4;
    atomicAdd(a + 0, m.x);
    atomicAdd(a + 1, m.y);
    atomicAdd(a + 2, m.z);
    atomicAdd(a + 3, m.w);
    if (lane == 0) atomicAdd(deg + dst, 1.0f);
}

// ---------------- Kernel 5: agg /= max(deg,1) and convert to bf16 ----------------
__global__ void div_deg_cvt(const float* __restrict__ agg, const float* __restrict__ deg,
                            unsigned short* __restrict__ aggb) {
    int i = blockIdx.x * blockDim.x + threadIdx.x;   // over N*D
    int row = i >> 7;
    float d = fmaxf(deg[row], 1.0f);
    aggb[i] = f32_to_bf16_rne(agg[i] / d);
}

// ---------------- Kernel 6: out = agg@Wl^T + b_l + h@Wr^T via bf16 WMMA ----------------
__global__ void __launch_bounds__(256)
gemm_wmma(const unsigned short* __restrict__ aggb, const unsigned short* __restrict__ hb,
          const unsigned short* __restrict__ wlb,  const unsigned short* __restrict__ wrb,
          const float* __restrict__ bl, float* __restrict__ out) {
    int wave = (blockIdx.x * blockDim.x + threadIdx.x) >> 5;
    int lane = threadIdx.x & 31;
    int tm = wave >> 3;            // node tile  0..3124
    int tn = wave & 7;             // out-feature tile 0..7
    int lrow = lane & 15;          // M (for A) / N (for B,C,D) within tile
    int hiK  = (lane >> 4) * 8;    // K sub-offset per ISA A/B 16-bit layout

    // Accumulator seeded with bias (broadcast over M)
    float bias = bl[tn * 16 + lrow];
    v8f c;
    #pragma unroll
    for (int r = 0; r < 8; ++r) c[r] = bias;

    const size_t arow = (size_t)(tm * 16 + lrow) * DFEAT;  // activation row
    const size_t brow = (size_t)(tn * 16 + lrow) * DFEAT;  // weight row == B column

    #pragma unroll
    for (int k0 = 0; k0 < DFEAT; k0 += 32) {
        // A = agg tile (16x32 bf16)
        v8bf alo = *(const v8bf*)(aggb + arow + k0 + hiK);
        v8bf ahi = *(const v8bf*)(aggb + arow + k0 + 16 + hiK);
        v16bf a = __builtin_shufflevector(alo, ahi, 0,1,2,3,4,5,6,7,8,9,10,11,12,13,14,15);
        // B = W_l tile (32x16 bf16), column = weight row
        v8bf blo = *(const v8bf*)(wlb + brow + k0 + hiK);
        v8bf bhi = *(const v8bf*)(wlb + brow + k0 + 16 + hiK);
        v16bf b = __builtin_shufflevector(blo, bhi, 0,1,2,3,4,5,6,7,8,9,10,11,12,13,14,15);
        c = __builtin_amdgcn_wmma_f32_16x16x32_bf16(false, a, false, b, (short)0, c, false, false);

        // A2 = h tile, B2 = W_r tile
        v8bf a2lo = *(const v8bf*)(hb + arow + k0 + hiK);
        v8bf a2hi = *(const v8bf*)(hb + arow + k0 + 16 + hiK);
        v16bf a2 = __builtin_shufflevector(a2lo, a2hi, 0,1,2,3,4,5,6,7,8,9,10,11,12,13,14,15);
        v8bf b2lo = *(const v8bf*)(wrb + brow + k0 + hiK);
        v8bf b2hi = *(const v8bf*)(wrb + brow + k0 + 16 + hiK);
        v16bf b2 = __builtin_shufflevector(b2lo, b2hi, 0,1,2,3,4,5,6,7,8,9,10,11,12,13,14,15);
        c = __builtin_amdgcn_wmma_f32_16x16x32_bf16(false, a2, false, b2, (short)0, c, false, false);
    }

    // D layout: VGPR r -> row r + (lane>=16 ? 8 : 0); lane&15 -> column
    int col   = tn * 16 + lrow;
    int rbase = tm * 16 + hiK;     // hiK == (lane>=16 ? 8 : 0)
    #pragma unroll
    for (int r = 0; r < 8; ++r)
        out[(size_t)(rbase + r) * DFEAT + col] = c[r];
}

extern "C" void kernel_launch(void* const* d_in, const int* in_sizes, int n_in,
                              void* d_out, int out_size, void* d_ws, size_t ws_size,
                              hipStream_t stream) {
    const float* x     = (const float*)d_in[0];
    const int*   ei    = (const int*)  d_in[1];
    const float* mask  = (const float*)d_in[2];
    const float* gamma = (const float*)d_in[3];
    const float* beta  = (const float*)d_in[4];
    const float* wl    = (const float*)d_in[5];
    const float* bl    = (const float*)d_in[6];
    const float* wr    = (const float*)d_in[7];
    float* out = (float*)d_out;

    // Workspace carve-up (all 256B aligned; agg+deg kept contiguous for one zero pass)
    char* ws = (char*)d_ws;
    size_t off = 0;
    auto carve = [&](size_t bytes) { void* p = ws + off; off = (off + bytes + 255) & ~(size_t)255; return p; };
    float*          h_f32  = (float*)carve((size_t)N_NODES * DFEAT * 4);
    float*          agg    = (float*)carve((size_t)N_NODES * DFEAT * 4 + (size_t)N_NODES * 4); // agg then deg
    float*          deg    = agg + (size_t)N_NODES * DFEAT;
    unsigned short* h_bf   = (unsigned short*)carve((size_t)N_NODES * DFEAT * 2);
    unsigned short* agg_bf = (unsigned short*)carve((size_t)N_NODES * DFEAT * 2);
    unsigned short* wl_bf  = (unsigned short*)carve((size_t)DFEAT * DFEAT * 2);
    unsigned short* wr_bf  = (unsigned short*)carve((size_t)DFEAT * DFEAT * 2);

    const int nzero = N_NODES * DFEAT + N_NODES;
    zero_f32<<<(nzero + 255) / 256, 256, 0, stream>>>(agg, nzero);
    cvt_f32_bf16<<<(DFEAT * DFEAT + 255) / 256, 256, 0, stream>>>(wl, wl_bf, DFEAT * DFEAT);
    cvt_f32_bf16<<<(DFEAT * DFEAT + 255) / 256, 256, 0, stream>>>(wr, wr_bf, DFEAT * DFEAT);

    // 8 waves per block, one wave per row: 50000/8 = 6250 blocks
    ln_relu_drop<<<N_NODES / 8, 256, 0, stream>>>(x, mask, gamma, beta, h_f32, h_bf);

    // one wave per edge: 800000/8 = 100000 blocks
    scatter_edges<<<E_EDGES / 8, 256, 0, stream>>>(ei, h_f32, agg, deg);

    // normalize + convert: 6.4M elements
    div_deg_cvt<<<(N_NODES * DFEAT) / 256, 256, 0, stream>>>(agg, deg, agg_bf);

    // WMMA epilogue: 3125 m-tiles x 8 n-tiles = 25000 waves = 3125 blocks of 8 waves
    gemm_wmma<<<(N_NODES / 16) * 8 / 8, 256, 0, stream>>>(agg_bf, h_bf, wl_bf, wr_bf, bl, out);
}
